// Attention_6966436954544
// MI455X (gfx1250) — matrix operand order repaired and checked by reading, not compiled, not run on hardware
//
#include <hip/hip_runtime.h>

// MI455X / gfx1250, wave32. Matmuls via v_wmma_f32_16x16x32_f16 (f32 accum),
// tile staging via global_load_async_to_lds_b128 (ASYNCcnt).
//
// B=8, N=1024, DIM=512, H=8, DH=64, MAX_POS=512, SCALE=1/8.

typedef __attribute__((ext_vector_type(16))) _Float16 v16h;
typedef __attribute__((ext_vector_type(8)))  float    v8f;

#define WMMA_F16(A,B,C) \
  __builtin_amdgcn_wmma_f32_16x16x32_f16(false,(A),false,(B),(short)0,(C),false,false)

// LDS byte offset of a __shared__ object: an addrspace(3) pointer's value is
// the offset within the wave's LDS allocation.
static __device__ __forceinline__ unsigned lds_off(const void* p) {
  return (unsigned)(unsigned long long)(const __attribute__((address_space(3))) char*)p;
}
// Async copy of 16 bytes/lane, global -> LDS. vdst = per-lane LDS byte addr,
// vaddr = per-lane 64-bit global address (GV mode).
static __device__ __forceinline__ void async_cp16(unsigned lds, const void* g) {
  asm volatile("global_load_async_to_lds_b128 %0, %1, off"
               :: "v"(lds), "v"(g) : "memory");
}
static __device__ __forceinline__ void wait_async() {
  asm volatile("s_wait_asynccnt 0x0" ::: "memory");
}

static __device__ __forceinline__ float redmax16(float v) {
#pragma unroll
  for (int m = 1; m < 16; m <<= 1) v = fmaxf(v, __shfl_xor(v, m, 32));
  return v;
}
static __device__ __forceinline__ float redsum16(float v) {
#pragma unroll
  for (int m = 1; m < 16; m <<= 1) v += __shfl_xor(v, m, 32);
  return v;
}

// -------------------------------------------------------------------------
// Kernel 0: E (1025 x 64 f32) -> Eh (1040 x 64 f16, rows >= 1025 zeroed).
// -------------------------------------------------------------------------
__global__ __launch_bounds__(256) void econv(const float* __restrict__ E,
                                             _Float16* __restrict__ Eh)
{
  int i = blockIdx.x * 256 + threadIdx.x;          // 1040*64 = 66560 elements
  if (i < 1040 * 64) {
    int row = i >> 6;
    Eh[i] = (row <= 1024) ? (_Float16)E[i] : (_Float16)0.0f;
  }
}

// -------------------------------------------------------------------------
// Kernel 1: fused QKV projection.
// rows = B*N = 8192 tokens, cols = 1536 (Q:0..511 | K:512..1023 | V:1024..1535)
// grid (512, 24), block 128 (4 waves, each owns one 16x16 output tile).
// Q is pre-scaled by SCALE (used by both dot and rel-pos terms).
// -------------------------------------------------------------------------
__global__ __launch_bounds__(128) void qkv_proj(
    const float* __restrict__ x, const float* __restrict__ Wq,
    const float* __restrict__ Wkv,
    _Float16* __restrict__ Qh, _Float16* __restrict__ Kh,
    _Float16* __restrict__ Vh)
{
  __shared__ _Float16 lA[16 * 32];
  __shared__ _Float16 lB[4][16 * 32];   // transposed: lB[w][col*32 + k]

  const int tid  = threadIdx.x;
  const int wave = tid >> 5;
  const int lane = tid & 31;
  const int nn   = lane & 15;
  const int kb   = (lane < 16) ? 0 : 8;
  const int kb2  = (lane < 16) ? 0 : 16;
  const int rb   = (lane < 16) ? 0 : 8;
  const int r0   = blockIdx.x * 16;
  const int c0   = blockIdx.y * 64 + wave * 16;

  const float* W; int ncols, wc0;
  if (c0 < 512) { W = Wq;  ncols = 512;  wc0 = c0; }
  else          { W = Wkv; ncols = 1024; wc0 = c0 - 512; }

  v8f acc = {};
  for (int ks = 0; ks < 16; ++ks) {
    const int k0 = ks * 32;
#pragma unroll
    for (int t = 0; t < 4; ++t) {                  // stage x tile (f32->f16)
      int e = tid + t * 128;
      int i = e >> 5, j = e & 31;
      lA[e] = (_Float16)x[(size_t)(r0 + i) * 512 + k0 + j];
    }
    {                                              // stage W tile transposed
      const float* wp = W + (size_t)(k0 + lane) * ncols + wc0;
#pragma unroll
      for (int t = 0; t < 16; ++t) lB[wave][t * 32 + lane] = (_Float16)wp[t];
    }
    __syncthreads();
    v16h a, b;
#pragma unroll
    for (int e = 0; e < 16; ++e) a[e] = lA[nn * 32 + kb2 + e];
#pragma unroll
    for (int e = 0; e < 16; ++e) b[e] = lB[wave][nn * 32 + kb2 + e];
    acc = WMMA_F16(a, b, acc);
    __syncthreads();
  }

  const int cg = c0 + nn;
  const float sc = (cg < 512) ? 0.125f : 1.0f;     // SCALE folded into Q
  _Float16* dst; int cc;
  if      (cg < 512)  { dst = Qh; cc = cg; }
  else if (cg < 1024) { dst = Kh; cc = cg - 512; }
  else                { dst = Vh; cc = cg - 1024; }
  const int hh = cc >> 6, dd = cc & 63;
#pragma unroll
  for (int j = 0; j < 8; ++j) {
    int row = r0 + j + rb;                         // token row in [0,8192)
    int bb = row >> 10, n = row & 1023;
    dst[(((size_t)bb * 8 + hh) * 1024 + n) * 64 + dd] = (_Float16)(acc[j] * sc);
  }
}

// -------------------------------------------------------------------------
// Kernel 2: relative-position flash attention.
// One wave per (b, h, 16-query-row block). grid (64, 8, 8), block 32.
// G = q @ E^T (16 x 1025) built once per block in LDS; pos term is a gather.
// All f16 tile staging uses async global->LDS copies (row-major layouts: K
// row-major in LDS == transposed-B layout, so K fragments read contiguously; V is read strided).
// -------------------------------------------------------------------------
__global__ __launch_bounds__(32) void rel_attn(
    const _Float16* __restrict__ Qh, const _Float16* __restrict__ Kh,
    const _Float16* __restrict__ Vh, const _Float16* __restrict__ Eh,
    _Float16* __restrict__ AO)
{
  __shared__ _Float16 lq[16 * 64];      // q block, row-major
  __shared__ _Float16 lG[16 * 1040];    // q @ E^T (f16)
  __shared__ _Float16 lK[32 * 64];      // 32-key K block, row-major
  __shared__ _Float16 lE[16 * 64];      // E tile, row-major
  __shared__ _Float16 lP[16 * 32];      // probabilities tile (f16)
  __shared__ _Float16 lV[32 * 64];      // 32-key V block, row-major

  const int lane = threadIdx.x;
  const int nn   = lane & 15;
  const int kb   = (lane < 16) ? 0 : 8;
  const int kb2  = (lane < 16) ? 0 : 16;
  const int rb   = (lane < 16) ? 0 : 8;
  const int qt = blockIdx.x, h = blockIdx.y, b = blockIdx.z;
  const int n0 = qt * 16;
  const size_t bh = (size_t)b * 8 + h;
  const _Float16* Qp = Qh + bh * 1024 * 64;
  const _Float16* Kp = Kh + bh * 1024 * 64;
  const _Float16* Vp = Vh + bh * 1024 * 64;

  // ---- q block -> LDS (16 rows x 128B = 4 async chunks/lane) ----
#pragma unroll
  for (int i = 0; i < 4; ++i) {
    int ch = lane + 32 * i, row = ch >> 3, seg = ch & 7;
    async_cp16(lds_off(&lq[row * 64 + seg * 8]),
               Qp + (size_t)(n0 + row) * 64 + seg * 8);
  }
  wait_async();
  __syncthreads();

  v16h aq0, aq1;                        // q fragments, d=0..31 / 32..63
#pragma unroll
  for (int e = 0; e < 16; ++e) aq0[e] = lq[nn * 64 + 0  + kb2 + e];
#pragma unroll
  for (int e = 0; e < 16; ++e) aq1[e] = lq[nn * 64 + 32 + kb2 + e];

  // ---- G = q @ Eh^T (16 x 1040, 65 col tiles) ----
  for (int jt = 0; jt < 65; ++jt) {
    const int j0 = jt * 16;
#pragma unroll
    for (int i = 0; i < 4; ++i) {
      int ch = lane + 32 * i, row = ch >> 3, seg = ch & 7;
      async_cp16(lds_off(&lE[row * 64 + seg * 8]),
                 Eh + (size_t)(j0 + row) * 64 + seg * 8);
    }
    wait_async();
    __syncthreads();
    v8f acc = {};
#pragma unroll
    for (int s = 0; s < 2; ++s) {
      v16h bf;                          // B[k][col] = Eh[j0+col][k]
#pragma unroll
      for (int e = 0; e < 16; ++e) bf[e] = lE[nn * 64 + s * 32 + kb2 + e];
      acc = WMMA_F16(s == 0 ? aq0 : aq1, bf, acc);
    }
#pragma unroll
    for (int j = 0; j < 8; ++j)
      lG[(j + rb) * 1040 + j0 + nn] = (_Float16)acc[j];
    __syncthreads();                    // frag reads done before lE restage
  }

  // ---- flash loop over 32-key blocks ----
  float rm[8], rl[8];
#pragma unroll
  for (int j = 0; j < 8; ++j) { rm[j] = -3.0e38f; rl[j] = 0.0f; }
  v8f O0 = {}, O1 = {}, O2 = {}, O3 = {};

  for (int kt = 0; kt < 32; ++kt) {
    const int kk0 = kt * 32;
    // stage K and V blocks (32 rows x 128B each = 8 async chunks/lane each)
#pragma unroll
    for (int i = 0; i < 8; ++i) {
      int ch = lane + 32 * i, row = ch >> 3, seg = ch & 7;
      async_cp16(lds_off(&lK[row * 64 + seg * 8]),
                 Kp + (size_t)(kk0 + row) * 64 + seg * 8);
    }
#pragma unroll
    for (int i = 0; i < 8; ++i) {
      int ch = lane + 32 * i, row = ch >> 3, seg = ch & 7;
      async_cp16(lds_off(&lV[row * 64 + seg * 8]),
                 Vp + (size_t)(kk0 + row) * 64 + seg * 8);
    }
    wait_async();
    __syncthreads();

    float sf0[8], sf1[8];
#pragma unroll
    for (int u = 0; u < 2; ++u) {       // two 16-key score tiles
      const int r0 = kk0 + u * 16;
      v8f acc = {};
#pragma unroll
      for (int s = 0; s < 2; ++s) {
        v16h bf;                        // B[k][col] = K[r0+col][k]
#pragma unroll
        for (int e = 0; e < 16; ++e)
          bf[e] = lK[(u * 16 + nn) * 64 + s * 32 + kb2 + e];
        acc = WMMA_F16(s == 0 ? aq0 : aq1, bf, acc);
      }
#pragma unroll
      for (int j = 0; j < 8; ++j) {     // add rel-pos via G gather
        int nq = n0 + j + rb;
        int rr = r0 + nn;
        int d  = nq - rr;
        d = (d < -512) ? -512 : (d > 512 ? 512 : d);
        float sv = acc[j] + (float)lG[(j + rb) * 1040 + d + 512];
        if (u == 0) sf0[j] = sv; else sf1[j] = sv;
      }
    }
    // online softmax update (per-row stats replicated across 16 lanes)
    float alpha[8];
#pragma unroll
    for (int j = 0; j < 8; ++j) {
      float tm   = redmax16(fmaxf(sf0[j], sf1[j]));
      float mnew = fmaxf(rm[j], tm);
      alpha[j]   = __expf(rm[j] - mnew);
      float p0   = __expf(sf0[j] - mnew);
      float p1   = __expf(sf1[j] - mnew);
      rl[j] = rl[j] * alpha[j] + redsum16(p0 + p1);
      rm[j] = mnew;
      lP[(j + rb) * 32 + nn]      = (_Float16)p0;
      lP[(j + rb) * 32 + 16 + nn] = (_Float16)p1;
    }
#pragma unroll
    for (int j = 0; j < 8; ++j) {
      O0[j] *= alpha[j]; O1[j] *= alpha[j]; O2[j] *= alpha[j]; O3[j] *= alpha[j];
    }
    __syncthreads();
    v16h ap;
#pragma unroll
    for (int e = 0; e < 16; ++e) ap[e] = lP[nn * 32 + kb2 + e];
    v16h bv0, bv1, bv2, bv3;            // B[k][c] = V[kk0+k][c], row-major LDS
#pragma unroll
    for (int e = 0; e < 16; ++e) bv0[e] = lV[(kb2 + e) * 64 +  0 + nn];
#pragma unroll
    for (int e = 0; e < 16; ++e) bv1[e] = lV[(kb2 + e) * 64 + 16 + nn];
#pragma unroll
    for (int e = 0; e < 16; ++e) bv2[e] = lV[(kb2 + e) * 64 + 32 + nn];
#pragma unroll
    for (int e = 0; e < 16; ++e) bv3[e] = lV[(kb2 + e) * 64 + 48 + nn];
    O0 = WMMA_F16(ap, bv0, O0);
    O1 = WMMA_F16(ap, bv1, O1);
    O2 = WMMA_F16(ap, bv2, O2);
    O3 = WMMA_F16(ap, bv3, O3);
    __syncthreads();                    // reads done before next restage
  }

#pragma unroll
  for (int j = 0; j < 8; ++j) {
    float inv = 1.0f / rl[j];
    int n = n0 + j + rb;
    _Float16* op = AO + ((size_t)b * 1024 + n) * 512 + h * 64;
    op[ 0 + nn] = (_Float16)(O0[j] * inv);
    op[16 + nn] = (_Float16)(O1[j] * inv);
    op[32 + nn] = (_Float16)(O2[j] * inv);
    op[48 + nn] = (_Float16)(O3[j] * inv);
  }
}

// -------------------------------------------------------------------------
// Kernel 3: output projection (8192x512)@(512x512) + bias, f32 out.
// grid (512, 8), block 128.
// -------------------------------------------------------------------------
__global__ __launch_bounds__(128) void out_proj(
    const _Float16* __restrict__ AO, const float* __restrict__ Wo,
    const float* __restrict__ bo, float* __restrict__ out)
{
  __shared__ _Float16 lA[16 * 32];
  __shared__ _Float16 lB[4][16 * 32];

  const int tid  = threadIdx.x;
  const int wave = tid >> 5;
  const int lane = tid & 31;
  const int nn   = lane & 15;
  const int kb   = (lane < 16) ? 0 : 8;
  const int kb2  = (lane < 16) ? 0 : 16;
  const int rb   = (lane < 16) ? 0 : 8;
  const int r0   = blockIdx.x * 16;
  const int c0   = blockIdx.y * 64 + wave * 16;

  v8f acc = {};
  for (int ks = 0; ks < 16; ++ks) {
    const int k0 = ks * 32;
#pragma unroll
    for (int t = 0; t < 4; ++t) {
      int e = tid + t * 128;
      int i = e >> 5, j = e & 31;
      lA[e] = AO[(size_t)(r0 + i) * 512 + k0 + j];
    }
    {
      const float* wp = Wo + (size_t)(k0 + lane) * 512 + c0;
#pragma unroll
      for (int t = 0; t < 16; ++t) lB[wave][t * 32 + lane] = (_Float16)wp[t];
    }
    __syncthreads();
    v16h a, b;
#pragma unroll
    for (int e = 0; e < 16; ++e) a[e] = lA[nn * 32 + kb2 + e];
#pragma unroll
    for (int e = 0; e < 16; ++e) b[e] = lB[wave][nn * 32 + kb2 + e];
    acc = WMMA_F16(a, b, acc);
    __syncthreads();
  }

  const float bias = bo[c0 + nn];
#pragma unroll
  for (int j = 0; j < 8; ++j)
    out[(size_t)(r0 + j + rb) * 512 + c0 + nn] = acc[j] + bias;
}

// -------------------------------------------------------------------------
extern "C" void kernel_launch(void* const* d_in, const int* in_sizes, int n_in,
                              void* d_out, int out_size, void* d_ws, size_t ws_size,
                              hipStream_t stream)
{
  (void)in_sizes; (void)n_in; (void)out_size; (void)ws_size;
  const float* x   = (const float*)d_in[0];
  const float* Wq  = (const float*)d_in[1];
  const float* Wkv = (const float*)d_in[2];
  const float* Wo  = (const float*)d_in[3];
  const float* bo  = (const float*)d_in[4];
  const float* E   = (const float*)d_in[5];
  float* out = (float*)d_out;

  const size_t QKV = (size_t)8 * 8 * 1024 * 64;    // 4,194,304 elems each
  _Float16* Qh = (_Float16*)d_ws;
  _Float16* Kh = Qh + QKV;
  _Float16* Vh = Kh + QKV;
  _Float16* AO = Vh + QKV;                         // [b][n][h*64+d]
  _Float16* Eh = AO + QKV;                         // 1040 x 64 f16

  econv   <<<dim3(260),       256, 0, stream>>>(E, Eh);
  qkv_proj<<<dim3(512, 24),   128, 0, stream>>>(x, Wq, Wkv, Qh, Kh, Vh);
  rel_attn<<<dim3(64, 8, 8),   32, 0, stream>>>(Qh, Kh, Vh, Eh, AO);
  out_proj<<<dim3(512, 8),    128, 0, stream>>>(AO, Wo, bo, out);
}
